// NonLocalGeneric_69063074120020
// MI455X (gfx1250) — matrix operand order repaired and checked by reading, not compiled, not run on hardware
//
#include <hip/hip_runtime.h>

#define B_  8
#define C_  256
#define N_  4096
#define CK_ 128
#define CV_ 128

typedef __attribute__((ext_vector_type(16))) __bf16 v16bf;
typedef __attribute__((ext_vector_type(8)))  float  v8f;

union Frag { unsigned u[8]; v16bf v; };

// RNE f32->bf16 pair pack (pre-pass / epilogue use)
__device__ __forceinline__ unsigned bfpack2(float a, float b) {
    unsigned ua = __builtin_bit_cast(unsigned, a);
    unsigned ub = __builtin_bit_cast(unsigned, b);
    ua = (ua + 0x7FFFu + ((ua >> 16) & 1u)) >> 16;
    ub = (ub + 0x7FFFu + ((ub >> 16) & 1u)) >> 16;
    return (ua & 0xFFFFu) | (ub << 16);
}
__device__ __forceinline__ unsigned short bf1(float a) {
    unsigned ua = __builtin_bit_cast(unsigned, a);
    return (unsigned short)((ua + 0x7FFFu + ((ua >> 16) & 1u)) >> 16);
}
// single-op truncating pack: low16=bf16(a), high16=bf16(b)  (v_perm_b32)
__device__ __forceinline__ unsigned permpack(float a, float b) {
    return __builtin_amdgcn_perm(__builtin_bit_cast(unsigned, b),
                                 __builtin_bit_cast(unsigned, a), 0x07060302u);
}
__device__ __forceinline__ Frag ld_frag16(const unsigned short* p, int second_off) {
    Frag f;
    uint4 a  = *reinterpret_cast<const uint4*>(p);
    uint4 b2 = *reinterpret_cast<const uint4*>(p + second_off);
    f.u[0] = a.x;  f.u[1] = a.y;  f.u[2] = a.z;  f.u[3] = a.w;
    f.u[4] = b2.x; f.u[5] = b2.y; f.u[6] = b2.z; f.u[7] = b2.w;
    return f;
}

#define WMMA_BF16(A, Bv, Cv) \
    __builtin_amdgcn_wmma_f32_16x16x32_bf16(false, (A), false, (Bv), (short)0, (Cv), false, false)

// ---------------------------------------------------------------------------
// Kernel 0a: pack x (f32 [B,C,N]) into xp (u32 [B,C/2,N]): dword = bf16 pair of
// channels (2c,2c+1) at one spatial position == B-fragment register image.
// ---------------------------------------------------------------------------
__global__ __launch_bounds__(256) void packx_kernel(
    const float* __restrict__ x, unsigned* __restrict__ xp)
{
    const size_t total = (size_t)B_ * (C_ / 2) * N_;
    size_t i = (size_t)blockIdx.x * blockDim.x + threadIdx.x;
    if (i >= total) return;
    size_t n  = i % N_;
    size_t c2 = (i / N_) % (C_ / 2);
    size_t b  = i / ((size_t)(C_ / 2) * N_);
    const float* px = x + b * C_ * N_ + (2 * c2) * N_ + n;
    xp[i] = bfpack2(px[0], px[N_]);
}

// Kernel 0b: convert a contiguous f32 array to packed bf16 (u32 holds 2 elems).
__global__ __launch_bounds__(256) void wpack_kernel(
    const float* __restrict__ in, unsigned* __restrict__ out, int npairs)
{
    int i = blockIdx.x * blockDim.x + threadIdx.x;
    if (i >= npairs) return;
    out[i] = bfpack2(in[2 * i], in[2 * i + 1]);
}

// ---------------------------------------------------------------------------
// Kernel 1: input projections.  out = W(128x256) @ x(256x4096) + bias.
//   mi=0 theta -> Q  [N,CK] bf16   mi=1 phi -> Kr [N,CK] bf16 (same path)
//   mi=2 g     -> Vt [CV,N] bf16 (spatial-contiguous rows)
// ---------------------------------------------------------------------------
__global__ __launch_bounds__(256) void proj_kernel(
    const unsigned* __restrict__ XP,
    const unsigned short* __restrict__ WPB,   // [3][CK, C] bf16
    const float* __restrict__ theta_b, const float* __restrict__ phi_b,
    const float* __restrict__ g_b,
    unsigned short* __restrict__ Q, unsigned short* __restrict__ KR,
    unsigned short* __restrict__ VT)
{
    const int b    = blockIdx.y;
    const int nb   = blockIdx.x * 64;
    const int mi   = blockIdx.z;
    const unsigned short* wb = WPB + (size_t)mi * CK_ * C_;
    const float* bs = (mi == 0) ? theta_b : ((mi == 1) ? phi_b : g_b);

    const int lane = threadIdx.x & 31;
    const int wid  = threadIdx.x >> 5;      // 0..7 -> output-channel strip
    const int l    = lane & 15;
    const int hi   = lane >> 4;
    const int cr   = wid * 16;

    const unsigned* xb = XP + (size_t)b * (C_ / 2) * N_;

    v8f acc[4];
    for (int t = 0; t < 4; ++t) acc[t] = (v8f)(0.0f);

    for (int kc = 0; kc < C_; kc += 32) {
        Frag af = ld_frag16(wb + (size_t)(cr + l) * C_ + kc + hi * 16, 8);
        for (int t = 0; t < 4; ++t) {
            Frag bfr;   // 8 coalesced dword loads from pre-packed xp
            const int n = nb + t * 16 + l;
            const unsigned* xc = xb + (size_t)(kc / 2 + hi * 8) * N_ + n;
            for (int j = 0; j < 8; ++j)
                bfr.u[j] = xc[(size_t)j * N_];
            acc[t] = WMMA_BF16(af.v, bfr.v, acc[t]);
        }
    }

    if (mi == 2) {          // Vt: direct [CV, N] bf16
        unsigned short* O = VT + (size_t)b * CV_ * N_;
        for (int t = 0; t < 4; ++t) {
            const int n = nb + t * 16 + l;
            for (int r = 0; r < 8; ++r) {
                const int cv = cr + hi * 8 + r;
                O[(size_t)cv * N_ + n] = bf1(acc[t][r] + bs[cv]);
            }
        }
    } else {                // Q / Kr: transposed [N, CK] bf16, packed b128 stores
        unsigned short* O = ((mi == 0) ? Q : KR) + (size_t)b * N_ * CK_;
        for (int t = 0; t < 4; ++t) {
            const int n = nb + t * 16 + l;
            uint4 pk;
            pk.x = bfpack2(acc[t][0] + bs[cr + hi * 8 + 0], acc[t][1] + bs[cr + hi * 8 + 1]);
            pk.y = bfpack2(acc[t][2] + bs[cr + hi * 8 + 2], acc[t][3] + bs[cr + hi * 8 + 3]);
            pk.z = bfpack2(acc[t][4] + bs[cr + hi * 8 + 4], acc[t][5] + bs[cr + hi * 8 + 5]);
            pk.w = bfpack2(acc[t][6] + bs[cr + hi * 8 + 6], acc[t][7] + bs[cr + hi * 8 + 7]);
            *reinterpret_cast<uint4*>(O + (size_t)n * CK_ + cr + hi * 8) = pk;
        }
    }
}

// ---------------------------------------------------------------------------
// Kernel 2: fused attention, online softmax.  S^T = K @ Q^T so the score
// D-fragment IS the P A-fragment image (no LDS transpose).  Each wave now owns
// 32 queries (2 groups) so every K *and* V fragment feeds 2x the WMMAs,
// halving L2 traffic.  Q B-fragments reloaded per chunk (WGP$-resident slice).
// ---------------------------------------------------------------------------
__global__ __launch_bounds__(128) void attn_kernel(
    const unsigned short* __restrict__ Q,
    const unsigned short* __restrict__ KR,
    const unsigned short* __restrict__ VT,
    unsigned short* __restrict__ Y)
{
    const int b    = blockIdx.y;
    const int lane = threadIdx.x & 31;
    const int wid  = threadIdx.x >> 5;          // 0..3
    const int l    = lane & 15;
    const int hi   = lane >> 4;
    const int q0   = blockIdx.x * 128 + wid * 32;   // 32 queries per wave

    const unsigned short* Qb = Q  + (size_t)b * N_ * CK_;
    const unsigned short* Kb = KR + (size_t)b * N_ * CK_;
    const unsigned short* Vb = VT + (size_t)b * CV_ * N_;

    v8f yacc[2][8];
    for (int g = 0; g < 2; ++g)
        for (int v = 0; v < 8; ++v) yacc[g][v] = (v8f)(0.0f);
    float rmax[2] = {-3.0e38f, -3.0e38f}, rsum[2] = {0.0f, 0.0f};
    const float LOG2E = 1.4426950408889634f;

    for (int m0 = 0; m0 < N_; m0 += 32) {
        if (m0 + 32 < N_) {   // stream hints (global_prefetch_b8)
            __builtin_prefetch(Kb + (size_t)(m0 + 32 + lane) * CK_, 0, 1);
            __builtin_prefetch(Vb + (size_t)lane * N_ + m0 + 32, 0, 1);
        }
        // ---- S^T chunk: rows = keys m0..m0+31, cols = 2 groups of 16 queries
        v8f st[4];   // [g*2 + mhalf]
        for (int i = 0; i < 4; ++i) st[i] = (v8f)(0.0f);
        for (int c = 0; c < 4; ++c) {
            Frag ka0 = ld_frag16(Kb + (size_t)(m0 + l)      * CK_ + c * 32 + hi * 8, 16);
            Frag ka1 = ld_frag16(Kb + (size_t)(m0 + 16 + l) * CK_ + c * 32 + hi * 8, 16);
            for (int g = 0; g < 2; ++g) {
                Frag qb = ld_frag16(
                    Qb + (size_t)(q0 + g * 16 + l) * CK_ + c * 32 + hi * 8, 16);
                st[g * 2 + 0] = WMMA_BF16(ka0.v, qb.v, st[g * 2 + 0]);
                st[g * 2 + 1] = WMMA_BF16(ka1.v, qb.v, st[g * 2 + 1]);
            }
        }
        // lane (l,hi): st[2g+h][r] = S[q = q0+16g+l][m0 + 16h + hi*8 + r]

        // ---- online softmax per group: lane-local reduce + one xor16 shuffle
        Frag pf[2];
        float corrb[2][8];
        for (int g = 0; g < 2; ++g) {
            v8f s0 = st[g * 2 + 0], s1 = st[g * 2 + 1];
            float mx = s0[0];
            for (int r = 1; r < 8; ++r) mx = fmaxf(mx, s0[r]);
            for (int r = 0; r < 8; ++r) mx = fmaxf(mx, s1[r]);
            mx = fmaxf(mx, __shfl_xor(mx, 16));
            float nm   = fmaxf(rmax[g], mx);
            float corr = exp2f((rmax[g] - nm) * LOG2E);
            rmax[g] = nm;
            float ssum = 0.0f;
            for (int r = 0; r < 8; ++r) { s0[r] = exp2f((s0[r] - nm) * LOG2E); ssum += s0[r]; }
            for (int r = 0; r < 8; ++r) { s1[r] = exp2f((s1[r] - nm) * LOG2E); ssum += s1[r]; }
            ssum += __shfl_xor(ssum, 16);
            rsum[g] = rsum[g] * corr + ssum;
            for (int r = 0; r < 8; ++r) corrb[g][r] = __shfl(corr, hi * 8 + r);
            // P A-fragment straight from S^T registers (one v_perm per pair)
            for (int j = 0; j < 4; ++j) {
                pf[g].u[j]     = permpack(s0[2 * j], s0[2 * j + 1]);
                pf[g].u[4 + j] = permpack(s1[2 * j], s1[2 * j + 1]);
            }
        }
        for (int g = 0; g < 2; ++g)
            for (int v = 0; v < 8; ++v)
                for (int r = 0; r < 8; ++r) yacc[g][v][r] *= corrb[g][r];

        // ---- Y += P @ V : each V fragment feeds both query groups ----
        for (int v = 0; v < 8; ++v) {
            Frag vf = ld_frag16(Vb + (size_t)(v * 16 + l) * N_ + m0 + hi * 8, 16);
            yacc[0][v] = WMMA_BF16(pf[0].v, vf.v, yacc[0][v]);
            yacc[1][v] = WMMA_BF16(pf[1].v, vf.v, yacc[1][v]);
        }
    }

    // ---- finalize: divide by row sums, store bf16 [N, CV] ----
    unsigned short* Yb = Y + (size_t)b * N_ * CV_;
    for (int g = 0; g < 2; ++g) {
        float inv = 1.0f / rsum[g];
        float invb[8];
        for (int r = 0; r < 8; ++r) invb[r] = __shfl(inv, hi * 8 + r);
        for (int v = 0; v < 8; ++v)
            for (int r = 0; r < 8; ++r)
                Yb[(size_t)(q0 + g * 16 + hi * 8 + r) * CV_ + v * 16 + l] =
                    bf1(yacc[g][v][r] * invb[r]);
    }
}

// ---------------------------------------------------------------------------
// Kernel 3: context = w_w(256x128) @ Y^T + w_b, stored f32 as [B, C, N].
// ---------------------------------------------------------------------------
__global__ __launch_bounds__(512) void outproj_kernel(
    const unsigned short* __restrict__ Y,
    const unsigned short* __restrict__ WOB,   // [C, CV] bf16
    const float* __restrict__ w_b,
    float* __restrict__ out)
{
    const int b    = blockIdx.y;
    const int nb   = blockIdx.x * 64;
    const int lane = threadIdx.x & 31;
    const int wid  = threadIdx.x >> 5;      // 0..15 -> C strip
    const int l    = lane & 15;
    const int hi   = lane >> 4;
    const int cr   = wid * 16;

    const unsigned short* Yb = Y + (size_t)b * N_ * CV_;

    v8f acc[4];
    for (int t = 0; t < 4; ++t) acc[t] = (v8f)(0.0f);

    for (int kc = 0; kc < CV_; kc += 32) {
        Frag af = ld_frag16(WOB + (size_t)(cr + l) * CV_ + kc + hi * 8, 16);
        for (int t = 0; t < 4; ++t) {
            const int n = nb + t * 16 + l;
            Frag bfr = ld_frag16(Yb + (size_t)n * CV_ + kc + hi * 8, 16);
            acc[t] = WMMA_BF16(af.v, bfr.v, acc[t]);
        }
    }

    float* ob = out + (size_t)b * C_ * N_;
    for (int t = 0; t < 4; ++t) {
        const int n = nb + t * 16 + l;
        for (int r = 0; r < 8; ++r) {
            const int c = cr + hi * 8 + r;
            ob[(size_t)c * N_ + n] = acc[t][r] + w_b[c];
        }
    }
}

// ---------------------------------------------------------------------------
extern "C" void kernel_launch(void* const* d_in, const int* in_sizes, int n_in,
                              void* d_out, int out_size, void* d_ws, size_t ws_size,
                              hipStream_t stream) {
    (void)in_sizes; (void)n_in; (void)out_size; (void)ws_size;
    const float* x       = (const float*)d_in[0];
    const float* theta_w = (const float*)d_in[1];
    const float* theta_b = (const float*)d_in[2];
    const float* phi_w   = (const float*)d_in[3];
    const float* phi_b   = (const float*)d_in[4];
    const float* g_w     = (const float*)d_in[5];
    const float* g_b     = (const float*)d_in[6];
    const float* w_w     = (const float*)d_in[7];
    const float* w_b     = (const float*)d_in[8];
    float* out = (float*)d_out;

    // workspace: xp | Q | Kr | Vt | Y | WPB | WOB   (~48.3 MB)
    unsigned*       XP  = (unsigned*)d_ws;                          // B*(C/2)*N u32
    unsigned short* Q   = (unsigned short*)(XP + (size_t)B_ * (C_ / 2) * N_);
    unsigned short* KR  = Q  + (size_t)B_ * N_ * CK_;
    unsigned short* VT  = KR + (size_t)B_ * N_ * CK_;
    unsigned short* Y   = VT + (size_t)B_ * CV_ * N_;
    unsigned short* WPB = Y  + (size_t)B_ * N_ * CV_;               // 3*CK*C bf16
    unsigned short* WOB = WPB + (size_t)3 * CK_ * C_;               // C*CV bf16

    const int wpairs = CK_ * C_ / 2;   // 16384 pairs per weight matrix
    wpack_kernel<<<(wpairs + 255) / 256, 256, 0, stream>>>(theta_w, (unsigned*)WPB, wpairs);
    wpack_kernel<<<(wpairs + 255) / 256, 256, 0, stream>>>(phi_w,   (unsigned*)(WPB + (size_t)CK_ * C_), wpairs);
    wpack_kernel<<<(wpairs + 255) / 256, 256, 0, stream>>>(g_w,     (unsigned*)(WPB + (size_t)2 * CK_ * C_), wpairs);
    wpack_kernel<<<(wpairs + 255) / 256, 256, 0, stream>>>(w_w,     (unsigned*)WOB, wpairs);

    const int packx_blocks = (int)(((size_t)B_ * (C_ / 2) * N_ + 255) / 256);
    packx_kernel<<<packx_blocks, 256, 0, stream>>>(x, XP);

    proj_kernel<<<dim3(N_ / 64, B_, 3), 256, 0, stream>>>(
        XP, WPB, theta_b, phi_b, g_b, Q, KR, VT);
    attn_kernel<<<dim3(N_ / 128, B_), 128, 0, stream>>>(Q, KR, VT, Y);
    outproj_kernel<<<dim3(N_ / 64, B_), 512, 0, stream>>>(Y, WOB, w_b, out);
}
